// VVAttention_84284438216882
// MI455X (gfx1250) — matrix operand order repaired and checked by reading, not compiled, not run on hardware
//
#include <hip/hip_runtime.h>
#include <stdint.h>

#define NH 12
#define NSEQ 1025
#define BATCH 8
#define DMODEL 768
#define HD 64
#define MROWS (BATCH * NSEQ)   /* 8200 GEMM rows */
#define QKV_OUT (3 * DMODEL)   /* 2304 */

typedef __attribute__((ext_vector_type(16))) __bf16 v16bf;
typedef __attribute__((ext_vector_type(2)))  __bf16 bf16x2;
typedef __attribute__((ext_vector_type(8)))  float  v8f;

// TDM: always emit on the device pass; builtin when available, inline asm
// otherwise (2-operand form, VADDR2/3 = NULL for <=2D tensors, ISA 7.3).
#if defined(__HIP_DEVICE_COMPILE__)
#define USE_TDM 1
#if __has_builtin(__builtin_amdgcn_tensor_load_to_lds)
#define USE_TDM_BUILTIN 1
#endif
#else
#define USE_TDM 0
#endif
#if defined(__has_include)
#if __has_include(<hip/amd_detail/amd_gfx1250_TDM.h>)
#define TDM_6ARG 1
#endif
#endif

typedef unsigned u32x4 __attribute__((ext_vector_type(4)));
typedef int i32x4 __attribute__((ext_vector_type(4)));
typedef int i32x8 __attribute__((ext_vector_type(8)));

// ---------- helpers ----------

__device__ __forceinline__ unsigned short f2bf(float f) {
  __bf16 h = (__bf16)f;                       // native v_cvt path on gfx1250
  return __builtin_bit_cast(unsigned short, h);
}

// two floats -> packed bf16x2 dword; maps to a single v_cvt_pk_bf16_f32
__device__ __forceinline__ unsigned pack_bf2(float a, float b) {
  bf16x2 p;
  p[0] = (__bf16)a;
  p[1] = (__bf16)b;
  return __builtin_bit_cast(unsigned, p);
}

union BFPack { uint4 u[2]; v16bf v; };

__device__ __forceinline__ v16bf pack2(uint4 a, uint4 b) {
  BFPack p; p.u[0] = a; p.u[1] = b; return p.v;
}

__device__ __forceinline__ v8f wmma_bf16(v16bf a, v16bf b, v8f c) {
  return __builtin_amdgcn_wmma_f32_16x16x32_bf16(false, a, false, b, (short)0, c,
                                                 false, false);
}

// A fragment (16x32 bf16, M x K), CDNA5 layout (ISA 7.12.2): lane<16 holds row
// lane, K {0..7,16..23}; lane>=16 holds row lane-16, K {8..15,24..31}.
__device__ __forceinline__ v16bf afrag16(const unsigned short* tile, int stride,
                                         int kbase) {
  int lane = threadIdx.x & 31;
  int mr = lane & 15, hf = lane >> 4;
  const unsigned short* r = tile + mr * stride + kbase + (hf ? 8 : 0);
  uint4 lo = *(const uint4*)r;
  uint4 hi = *(const uint4*)(r + 16);
  return pack2(lo, hi);
}

// B fragment (32x16 bf16, K x N); tile row n holds K-contiguous data of output
// column (nbase+n). lane<16: K {0..7,16..23} of column lane; lane>=16: K {8..15,24..31}.
__device__ __forceinline__ v16bf bfrag16(const unsigned short* tile, int stride,
                                         int nbase, int kbase) {
  int lane = threadIdx.x & 31;
  int ln = lane & 15, hf = lane >> 4;
  const unsigned short* r = tile + (nbase + ln) * stride + kbase + hf * 8;
  uint4 lo = *(const uint4*)r;
  uint4 hi = *(const uint4*)(r + 16);
  return pack2(lo, hi);
}

__device__ __forceinline__ void tdm_issue(u32x4 g0, i32x8 g1) {
#if USE_TDM
#if defined(USE_TDM_BUILTIN)
  i32x4 z = {0, 0, 0, 0};
#ifdef TDM_6ARG
  i32x8 z8 = {0, 0, 0, 0, 0, 0, 0, 0};
  __builtin_amdgcn_tensor_load_to_lds(g0, g1, z, z, z8, 0);
#else
  __builtin_amdgcn_tensor_load_to_lds(g0, g1, z, z, 0);
#endif
#else
  // D# group0 in an SGPR quad, group1 in an SGPR octet; groups 2/3 NULL.
  asm volatile("tensor_load_to_lds %0, %1" ::"s"(g0), "s"(g1) : "memory");
#endif
#else
  (void)g0; (void)g1;
#endif
}

__device__ __forceinline__ void tdm_wait0() {
#if USE_TDM
#if __has_builtin(__builtin_amdgcn_s_wait_tensorcnt)
  __builtin_amdgcn_s_wait_tensorcnt(0);
#else
  asm volatile("s_wait_tensorcnt 0x0" ::: "memory");
#endif
#endif
}

// Issue a 2D bf16 tile load via the Tensor Data Mover: tile1 rows x dim0 cols,
// rows_rem rows valid (OOB rows read as zero -> tail masking for free), LDS
// destination padded pad_amount+1 DWORDs every 2^(pad_interval+1) DWORDs.
__device__ __forceinline__ void tdm_load_2d_bf16(unsigned lds_addr,
                                                 const void* gaddr,
                                                 unsigned dim0,
                                                 unsigned rows_rem,
                                                 unsigned tile0, unsigned tile1,
                                                 unsigned stride0,
                                                 unsigned pad_interval,
                                                 unsigned pad_amount) {
  unsigned long long ga = (unsigned long long)(uintptr_t)gaddr;
  u32x4 g0;
  g0[0] = 1u;                                    // count=1, user descriptor
  g0[1] = lds_addr;                              // LDS byte address
  g0[2] = (unsigned)(ga & 0xffffffffu);          // global addr [31:0]
  g0[3] = (unsigned)((ga >> 32) & 0x01ffffffu) | (2u << 30);  // [56:32], type=2
  i32x8 g1;
  unsigned w0 = (1u << 16)                       // data_size = 2 bytes
                | (1u << 20)                     // pad_enable
                | (pad_interval << 22) | (pad_amount << 25);
  g1[0] = (int)w0;
  g1[1] = (int)((dim0 & 0xffffu) << 16);                     // tensor_dim0 lo16
  g1[2] = (int)((dim0 >> 16) | ((rows_rem & 0xffffu) << 16));// dim0 hi/dim1 lo
  g1[3] = (int)((rows_rem >> 16) | (tile0 << 16));           // dim1 hi, tile0
  g1[4] = (int)(tile1 & 0xffffu);                            // tile1, tile2=0
  g1[5] = (int)stride0;                                      // dim0 stride lo32
  g1[6] = 0;
  g1[7] = 0;
  tdm_issue(g0, g1);
}

// ---------- kernel 1/5: fused bf16 WMMA GEMM (QKV proj and output proj) ----------
__global__ __launch_bounds__(256) void fused_gemm_kernel(
    int mode, const float* __restrict__ A, const float* __restrict__ W,
    const float* __restrict__ bias, float* __restrict__ q0,
    float* __restrict__ Kf, float* __restrict__ Vf, float* __restrict__ outp) {
  __shared__ alignas(16) unsigned short As[128 * 40];  // rows x (32 + 8 pad)
  __shared__ alignas(16) unsigned short Bs[64 * 40];

  const int tid = threadIdx.x;
  const int w = tid >> 5;
  const int row0 = blockIdx.x * 128;
  const int c0 = blockIdx.y * 64;

  v8f acc[4] = {v8f{}, v8f{}, v8f{}, v8f{}};

  for (int k0 = 0; k0 < DMODEL; k0 += 32) {
    __syncthreads();  // previous iteration's fragment reads done
    // stage A tile (128x32 f32 -> bf16)
#pragma unroll
    for (int j = 0; j < 4; ++j) {
      int c = tid + j * 256;            // 1024 chunks of 4
      int row = c >> 3, kk = (c & 7) << 2;
      int R = row0 + row;
      uint2 packed = {0u, 0u};
      if (R < MROWS) {
        const float* src;
        if (mode == 0) {
          int b = R / NSEQ, n = R % NSEQ;
          src = A + ((size_t)n * BATCH + b) * DMODEL + k0 + kk;
        } else {
          src = A + (size_t)R * DMODEL + k0 + kk;
        }
        if (k0 + 32 < DMODEL) __builtin_prefetch(src + 32, 0, 1);
        float4 f = *(const float4*)src;
        packed.x = pack_bf2(f.x, f.y);
        packed.y = pack_bf2(f.z, f.w);
      }
      *(uint2*)(As + row * 40 + kk) = packed;
    }
    // stage B tile (64 output cols x 32 k)
#pragma unroll
    for (int j = 0; j < 2; ++j) {
      int c = tid + j * 256;            // 512 chunks of 4
      int row = c >> 3, kk = (c & 7) << 2;
      const float* src = W + (size_t)(c0 + row) * DMODEL + k0 + kk;
      if (k0 + 32 < DMODEL) __builtin_prefetch(src + 32, 0, 1);
      float4 f = *(const float4*)src;
      uint2 packed;
      packed.x = pack_bf2(f.x, f.y);
      packed.y = pack_bf2(f.z, f.w);
      *(uint2*)(Bs + row * 40 + kk) = packed;
    }
    __syncthreads();

    v16bf a = afrag16(As + (w * 16) * 40, 40, 0);
#pragma unroll
    for (int j = 0; j < 4; ++j) {
      v16bf b = bfrag16(Bs, 40, j * 16, 0);
      acc[j] = wmma_bf16(a, b, acc[j]);
    }
  }

  // epilogue: C layout -> row = g + 8*(lane>=16), col = lane%16
  const int lane = tid & 31, ln = lane & 15, hf = lane >> 4;
#pragma unroll
  for (int j = 0; j < 4; ++j) {
#pragma unroll
    for (int g = 0; g < 8; ++g) {
      int R = row0 + w * 16 + g + 8 * hf;
      if (R >= MROWS) continue;
      float v = acc[j][g];
      int c = c0 + j * 16 + ln;
      int b = R / NSEQ, n = R % NSEQ;
      if (mode == 0) {
        if (c < DMODEL) {
          if (n == 0) q0[b * DMODEL + c] = v;  // only row 0 of Q survives
        } else if (c < 2 * DMODEL) {
          int c2 = c - DMODEL, hh = c2 >> 6, d = c2 & 63;
          Kf[(((size_t)(b * NH + hh)) * NSEQ + n) * HD + d] = v;
        } else {
          int c2 = c - 2 * DMODEL, hh = c2 >> 6, d = c2 & 63;
          Vf[(((size_t)(b * NH + hh)) * NSEQ + n) * HD + d] = v;
        }
      } else {
        outp[((size_t)n * BATCH + b) * DMODEL + c] = v + bias[c];
      }
    }
  }
}

// ---------- kernel 2/5: V row normalization, emit bf16 V and VN ----------
__global__ __launch_bounds__(256) void norm_kernel(
    const float* __restrict__ Vf, unsigned short* __restrict__ Vbf,
    unsigned short* __restrict__ VNbf) {
  int w = threadIdx.x >> 5, lane = threadIdx.x & 31;
  size_t row = (size_t)blockIdx.x * 8 + w;  // one 64-wide row per wave
  const float* src = Vf + row * HD;
  float a = src[lane], b = src[lane + 32];
  float ss = a * a + b * b;
#pragma unroll
  for (int off = 16; off; off >>= 1) ss += __shfl_xor(ss, off, 32);
  float inv = 1.0f / fmaxf(sqrtf(ss), 1e-6f);
  Vbf[row * HD + lane] = f2bf(a);
  Vbf[row * HD + lane + 32] = f2bf(b);
  VNbf[row * HD + lane] = f2bf(a * inv);
  VNbf[row * HD + lane + 32] = f2bf(b * inv);
}

// ---------- kernel 3/5: row-0 QK softmax path (tiny, f32 VALU) ----------
__global__ __launch_bounds__(128) void qk_row0_kernel(
    const float* __restrict__ q0, const float* __restrict__ Kf,
    const float* __restrict__ Vf, float* __restrict__ attn_out) {
  __shared__ float sc[1056];
  __shared__ float red[128];
  int tid = threadIdx.x, bh = blockIdx.x;
  int b = bh / NH, h = bh % NH;
  const float* q = q0 + b * DMODEL + h * HD;
  const float* Kp = Kf + (size_t)bh * NSEQ * HD;
  const float* Vp = Vf + (size_t)bh * NSEQ * HD;
  float lmax = -INFINITY;
  for (int n = tid; n < NSEQ; n += 128) {
    float s = 0.f;
#pragma unroll 8
    for (int d = 0; d < HD; ++d) s += q[d] * Kp[(size_t)n * HD + d];
    s *= 0.125f;
    sc[n] = s;
    lmax = fmaxf(lmax, s);
  }
  red[tid] = lmax; __syncthreads();
  for (int st = 64; st; st >>= 1) {
    if (tid < st) red[tid] = fmaxf(red[tid], red[tid + st]);
    __syncthreads();
  }
  float gmax = red[0]; __syncthreads();
  float lsum = 0.f;
  for (int n = tid; n < NSEQ; n += 128) {
    float e = __expf(sc[n] - gmax);
    sc[n] = e;
    lsum += e;
  }
  red[tid] = lsum; __syncthreads();
  for (int st = 64; st; st >>= 1) {
    if (tid < st) red[tid] += red[tid + st];
    __syncthreads();
  }
  float gsum = red[0]; __syncthreads();
  int d = tid & 63, hf = tid >> 6;
  float acc = 0.f;
  for (int n = hf; n < NSEQ; n += 2) acc += sc[n] * Vp[(size_t)n * HD + d];
  red[tid] = acc; __syncthreads();
  if (tid < 64)
    attn_out[((size_t)b * NSEQ) * DMODEL + h * HD + d] =
        (red[tid] + red[tid + 64]) / gsum;
}

// ---------- kernel 4/5: VV cosine flash attention (rows 1..N-1) ----------
// grid = B*H*16 blocks, 4 waves; wave owns 16 query rows; streams 32-key
// blocks through LDS (VN tile via TDM DMA) with online softmax.
__global__ __launch_bounds__(128) void vv_attn_kernel(
    const unsigned short* __restrict__ VNbf,
    const unsigned short* __restrict__ Vbf, float* __restrict__ attn_out) {
  __shared__ alignas(16) unsigned short VNt[32 * 72];     // [key][d], padded
  __shared__ alignas(16) unsigned short Vt[64 * 40];      // [d][key], padded
  __shared__ alignas(16) unsigned short Pt[4][16 * 40];   // per-wave P tile

  const int tid = threadIdx.x, w = tid >> 5;
  const int lane = tid & 31, ln = lane & 15, hf = lane >> 4;
  const int bh = blockIdx.x >> 4, qb = blockIdx.x & 15;
  const int b = bh / NH, h = bh % NH;
  const size_t base = (size_t)bh * NSEQ * HD;
  const int n0 = 1 + qb * 64 + w * 16;  // first query row of this wave
  const unsigned vnt_lds = (unsigned)(uintptr_t)(&VNt[0]);

  // Q fragments (normalized V rows) live in registers for the whole loop
  const unsigned short* qrow = VNbf + base + (size_t)n0 * HD;
  v16bf aq0 = afrag16(qrow, HD, 0);
  v16bf aq1 = afrag16(qrow, HD, 32);

  v8f o[4] = {v8f{}, v8f{}, v8f{}, v8f{}};
  float mrow[8], lrow[8];
#pragma unroll
  for (int g = 0; g < 8; ++g) { mrow[g] = -INFINITY; lrow[g] = 0.f; }

  const int NKB = (NSEQ + 31) / 32;  // 33
  for (int kb = 0; kb < NKB; ++kb) {
    const int key0 = kb * 32;
    __syncthreads();
#if USE_TDM
    // VN key tile [32][64] via Tensor Data Mover; wave 0 issues the DMA.
    // pad_interval=4 (32 DWORDs = one 64-bf16 row), pad_amount=3 (4 DWORDs)
    // reproduces the 72-ushort padded row stride; rows past NSEQ read zero.
    if (w == 0) {
      tdm_load_2d_bf16(vnt_lds, VNbf + base + (size_t)key0 * HD,
                       /*dim0=*/HD, /*rows_rem=*/(unsigned)(NSEQ - key0),
                       /*tile0=*/HD, /*tile1=*/32, /*stride0=*/HD,
                       /*pad_interval=*/4, /*pad_amount=*/3);
    }
#else
#pragma unroll
    for (int j = 0; j < 2; ++j) {
      int c = tid + j * 128;                    // 256 16B chunks
      int key = c >> 3, dch = (c & 7) << 3;
      uint4 v = {0u, 0u, 0u, 0u};
      if (key0 + key < NSEQ)
        v = *(const uint4*)(VNbf + base + (size_t)(key0 + key) * HD + dch);
      *(uint4*)(VNt + key * 72 + dch) = v;
    }
#endif
    // stage V key tile transposed [d][key] (B operand of P*V)
#pragma unroll
    for (int j = 0; j < 2; ++j) {
      int c = tid + j * 128;
      int key = c >> 3, dch = (c & 7) << 3;
      uint4 v = {0u, 0u, 0u, 0u};
      if (key0 + key < NSEQ)
        v = *(const uint4*)(Vbf + base + (size_t)(key0 + key) * HD + dch);
      const unsigned short* pe = (const unsigned short*)&v;
#pragma unroll
      for (int ii = 0; ii < 8; ++ii) Vt[(dch + ii) * 40 + key] = pe[ii];
    }
#if USE_TDM
    if (w == 0) tdm_wait0();
#endif
    __syncthreads();

    // S = VN_q . VN_k^T  (two 16-key tiles, K-dim = hd in two 32-chunks)
    v8f s[2] = {v8f{}, v8f{}};
#pragma unroll
    for (int t = 0; t < 2; ++t) {
      v16bf blo = bfrag16(VNt, 72, t * 16, 0);
      v16bf bhi = bfrag16(VNt, 72, t * 16, 32);
      s[t] = wmma_bf16(aq0, blo, s[t]);
      s[t] = wmma_bf16(aq1, bhi, s[t]);
    }

    // scale + key mask + per-row tile max
    float tmax[8];
#pragma unroll
    for (int g = 0; g < 8; ++g) tmax[g] = -INFINITY;
#pragma unroll
    for (int t = 0; t < 2; ++t) {
      bool ok = (key0 + t * 16 + ln) < NSEQ;
#pragma unroll
      for (int g = 0; g < 8; ++g) {
        float v = s[t][g] * 0.125f;
        if (!ok) v = -1e30f;
        s[t][g] = v;
        tmax[g] = fmaxf(tmax[g], v);
      }
    }
#pragma unroll
    for (int g = 0; g < 8; ++g) {
#pragma unroll
      for (int off = 8; off; off >>= 1)
        tmax[g] = fmaxf(tmax[g], __shfl_xor(tmax[g], off, 16));
    }
    // online softmax rescale
    float psum[8];
#pragma unroll
    for (int g = 0; g < 8; ++g) {
      float mnew = fmaxf(mrow[g], tmax[g]);
      float alpha = __expf(mrow[g] - mnew);
      mrow[g] = mnew;
      lrow[g] *= alpha;
      psum[g] = 0.f;
#pragma unroll
      for (int j = 0; j < 4; ++j) o[j][g] *= alpha;
    }
    // P = exp(S - m); stash in per-wave LDS to convert C-layout -> A-layout
    unsigned short* pt = Pt[w];
#pragma unroll
    for (int t = 0; t < 2; ++t) {
#pragma unroll
      for (int g = 0; g < 8; ++g) {
        float p = __expf(s[t][g] - mrow[g]);
        psum[g] += p;
        pt[(g + 8 * hf) * 40 + t * 16 + ln] = f2bf(p);
      }
    }
#pragma unroll
    for (int g = 0; g < 8; ++g) {
#pragma unroll
      for (int off = 8; off; off >>= 1) psum[g] += __shfl_xor(psum[g], off, 16);
      lrow[g] += psum[g];
    }
    // O += P . V   (K-dim = 32 keys, 4 d-tiles)
    v16bf ap = afrag16(pt, 40, 0);
#pragma unroll
    for (int j = 0; j < 4; ++j) {
      v16bf bv = bfrag16(Vt, 40, j * 16, 0);
      o[j] = wmma_bf16(ap, bv, o[j]);
    }
  }

  // epilogue: normalize by softmax sum, scatter into attn_out [b*N+n][D]
#pragma unroll
  for (int j = 0; j < 4; ++j) {
#pragma unroll
    for (int g = 0; g < 8; ++g) {
      int n = n0 + g + 8 * hf;
      attn_out[((size_t)b * NSEQ + n) * DMODEL + h * HD + j * 16 + ln] =
          o[j][g] / lrow[g];
    }
  }
}

// ---------- launch ----------
extern "C" void kernel_launch(void* const* d_in, const int* in_sizes, int n_in,
                              void* d_out, int out_size, void* d_ws,
                              size_t ws_size, hipStream_t stream) {
  (void)in_sizes; (void)n_in; (void)out_size; (void)ws_size;
  const float* x = (const float*)d_in[0];
  const float* w_qkv = (const float*)d_in[1];
  const float* w_proj = (const float*)d_in[2];
  const float* b_proj = (const float*)d_in[3];
  float* out = (float*)d_out;

  char* ws = (char*)d_ws;
  size_t off = 0;
  auto take = [&](size_t bytes) {
    char* p = ws + off;
    off += (bytes + 255) & ~(size_t)255;
    return p;
  };
  const size_t elems = (size_t)BATCH * NH * NSEQ * HD;  // 6.3M
  float* Kf = (float*)take(elems * 4);
  float* Vf = (float*)take(elems * 4);
  unsigned short* Vbf = (unsigned short*)take(elems * 2);
  unsigned short* VNbf = (unsigned short*)take(elems * 2);
  float* q0 = (float*)take((size_t)BATCH * DMODEL * 4);
  float* attn = (float*)take((size_t)MROWS * DMODEL * 4);

  // 1) QKV projection GEMM (bf16 WMMA) -> q0 / K / V
  dim3 g0((MROWS + 127) / 128, QKV_OUT / 64);
  fused_gemm_kernel<<<g0, 256, 0, stream>>>(0, x, w_qkv, nullptr, q0, Kf, Vf,
                                            nullptr);
  // 2) V row norms -> bf16 V, VN
  norm_kernel<<<(BATCH * NH * NSEQ) / 8, 256, 0, stream>>>(Vf, Vbf, VNbf);
  // 3) row-0 QK path
  qk_row0_kernel<<<BATCH * NH, 128, 0, stream>>>(q0, Kf, Vf, attn);
  // 4) VV flash attention for rows 1..1024 (bf16 WMMA + TDM staging)
  vv_attn_kernel<<<BATCH * NH * 16, 128, 0, stream>>>(VNbf, Vbf, attn);
  // 5) output projection GEMM + bias (bf16 WMMA) -> [N,B,D]
  dim3 g1((MROWS + 127) / 128, DMODEL / 64);
  fused_gemm_kernel<<<g1, 256, 0, stream>>>(1, attn, w_proj, b_proj, nullptr,
                                            nullptr, nullptr, out);
}
